// QuantumGQA_47107201303045
// MI455X (gfx1250) — matrix operand order repaired and checked by reading, not compiled, not run on hardware
//
#include <hip/hip_runtime.h>

// ---------------- problem constants ----------------
#define B_ 4
#define S_ 2048
#define E_ 1024
#define H_ 16
#define KV_ 4
#define HD_ 64
#define R_ 8
#define CD_ 64
#define NQ_ 8
#define NL_ 2
#define M_ (B_*S_)            // 8192 rows
#define N1_ 1600              // ctx(64) + qa(512) + ka(512) + va(512)
#define PI_F 3.14159265358979f

typedef __attribute__((ext_vector_type(2)))  float  v2f;
typedef __attribute__((ext_vector_type(8)))  float  v8f;
typedef __attribute__((ext_vector_type(8)))  __bf16 v8bf;
typedef __attribute__((ext_vector_type(16))) __bf16 v16bf;
typedef __attribute__((ext_vector_type(4)))  unsigned int u32x4;
typedef __attribute__((ext_vector_type(8)))  int i32x8;
typedef __attribute__((ext_vector_type(4)))  int i32x4;
typedef unsigned short ushort_t;
typedef unsigned int   uint_t;

#if defined(__has_builtin)
#  if __has_builtin(__builtin_amdgcn_tensor_load_to_lds) && __has_builtin(__builtin_amdgcn_s_wait_tensorcnt)
#    define HAVE_TDM 1
#  endif
#endif
#ifndef HAVE_TDM
#  define HAVE_TDM 0
#endif

__device__ __forceinline__ ushort_t f2bf(float f) {
  uint_t u = __builtin_bit_cast(uint_t, f);
  uint_t r = (u + 0x7FFFu + ((u >> 16) & 1u)) >> 16;
  return (ushort_t)r;
}

// ---------------- pack / convert kernels ----------------
__global__ void cvt_bf16_kernel(const float* __restrict__ src, ushort_t* __restrict__ dst, int n) {
  int i = blockIdx.x * 256 + threadIdx.x;
  if (i < n) dst[i] = f2bf(src[i]);
}

// copy [K x Ns] f32 into bf16 [K x Nd] at column offset
__global__ void cvt_cols_kernel(const float* __restrict__ src, ushort_t* __restrict__ dst,
                                int K, int Ns, int Nd, int off) {
  int i = blockIdx.x * 256 + threadIdx.x;
  if (i < K * Ns) {
    int k = i / Ns, n = i % Ns;
    dst[(size_t)k * Nd + off + n] = f2bf(src[i]);
  }
}

__global__ void pack_bias1_kernel(const float* __restrict__ ctxb, const float* __restrict__ qab,
                                  const float* __restrict__ kab, const float* __restrict__ vab,
                                  float* __restrict__ bias1) {
  int n = blockIdx.x * 256 + threadIdx.x;
  if (n < N1_) {
    float v;
    if      (n < 64)   v = ctxb[n];
    else if (n < 576)  v = qab[n - 64];
    else if (n < 1088) v = kab[n - 576];
    else               v = vab[n - 1088];
    bias1[n] = v;
  }
}

// ---------------- bf16 WMMA GEMM: C[M,N] = A[M,K](bf16) @ B[K,N](bf16) + bias[n] ----
// WG: 256 thr = 8 waves (2x4), wave tile 64x64, WG tile 128x256, K-step 32.
// A tile staged by the Tensor Data Mover (TDM) with hardware LDS pitch-padding;
// B tile staged manually because it must be transposed into [n][k].
__global__ __launch_bounds__(256) void gemm_bf16_kernel(
    const ushort_t* __restrict__ A, const ushort_t* __restrict__ B,
    const float* __restrict__ bias, float* __restrict__ C, int M, int N, int K) {
  __shared__ ushort_t sA[128][40];   // [m][k], pitch 40 ushorts = 64B row + 16B pad
  __shared__ ushort_t sB[256][40];   // transposed [n][k], pad 8
  int tid = threadIdx.x;
  int wave = tid >> 5, lane = tid & 31;
  int l16 = lane & 15, hi = lane >> 4;
  int wm = wave >> 2, wn = wave & 3;
  int m0 = blockIdx.x * 128;
  int n0 = blockIdx.y * 256;
  v8f acc[4][4] = {};
#if HAVE_TDM
  unsigned lds_a = (unsigned)(uintptr_t)(void*)&sA[0][0];  // flat addr low 32b == LDS offset
#endif

  for (int k0 = 0; k0 < K; k0 += 32) {
#if HAVE_TDM
    // ---- A tile via TDM: 2D tile (x=32, y=128) of bf16 out of [M,K] row-major ----
    if (wave == 0) {
      unsigned long long ga = (unsigned long long)(uintptr_t)(A + (size_t)m0 * K + k0);
      u32x4 g0;
      g0[0] = 1u;                                            // count=1, user descriptor
      g0[1] = lds_a;                                         // lds_addr
      g0[2] = (unsigned)(ga & 0xffffffffu);                  // global_addr[31:0]
      g0[3] = (unsigned)((ga >> 32) & 0x1ffffffu) | (2u << 30); // addr[56:32] | type=2
      i32x8 g1;
      // data_size=1(2B) | pad_enable | pad_interval=16DW(code 3) | pad_amount=4DW(code 3)
      g1[0] = (int)((1u << 16) | (1u << 20) | (3u << 22) | (3u << 25));
      g1[1] = (int)(((unsigned)K & 0xffffu) << 16);          // tensor_dim0[15:0]
      g1[2] = (int)((((unsigned)K >> 16) & 0xffffu) | (((unsigned)M & 0xffffu) << 16));
      g1[3] = (int)((((unsigned)M >> 16) & 0xffffu) | (32u << 16)); // tile_dim0=32
      g1[4] = (int)128u;                                     // tile_dim1=128, tile_dim2=0
      g1[5] = (int)(unsigned)K;                              // tensor_dim0_stride[31:0]
      g1[6] = 0;                                             // stride0 hi | stride1 lo
      g1[7] = 0;
      i32x4 gz4 = {0, 0, 0, 0};
      i32x8 gz8 = {0, 0, 0, 0, 0, 0, 0, 0};
      __builtin_amdgcn_tensor_load_to_lds(g0, g1, gz4, gz4, gz8, 0);
      __builtin_amdgcn_s_wait_tensorcnt(0);
    }
#else
    // manual A staging: 128x32 bf16 as 512 16B chunks
#pragma unroll
    for (int it = 0; it < 2; ++it) {
      int chunk = tid + it * 256;
      int row = chunk >> 2;
      int kk  = (chunk & 3) << 3;
      const uint4* src = (const uint4*)(A + (size_t)(m0 + row) * K + k0 + kk);
      *(uint4*)&sA[row][kk] = *src;
    }
#endif
    // stage B transposed: read rows of B coalesced, scatter to sB[n][k]
#pragma unroll
    for (int it = 0; it < 4; ++it) {
      int idx = tid + it * 256;
      int nn = (idx & 31) << 3;   // 0..248
      int kk = idx >> 5;          // 0..31
      union { uint4 u; ushort_t e[8]; } cvt;
      cvt.u = uint4{0u, 0u, 0u, 0u};
      if (n0 + nn < N) cvt.u = *(const uint4*)(B + (size_t)(k0 + kk) * N + n0 + nn);
#pragma unroll
      for (int j = 0; j < 8; ++j) sB[nn + j][kk] = cvt.e[j];
    }
    if (k0 + 32 < K) {
      __builtin_prefetch(B + (size_t)(k0 + 32 + (tid >> 5)) * N + n0 + ((tid & 31) << 3), 0, 0);
    }
    __syncthreads();

    // build fragments (ISA 7.12.2 16-bit A 16x32 layout; B staged as B^T)
    v16bf af[4], bfr[4];
#pragma unroll
    for (int t = 0; t < 4; ++t) {
      const ushort_t* pa = &sA[wm * 64 + t * 16 + l16][hi * 8];
      v8bf a_lo = *(const v8bf*)pa;          // K = 8*hi + 0..7
      v8bf a_hi = *(const v8bf*)(pa + 16);   // K = 16 + 8*hi + 0..7
#pragma unroll
      for (int j = 0; j < 8; ++j) { af[t][j] = a_lo[j]; af[t][8 + j] = a_hi[j]; }
      const ushort_t* pb = &sB[wn * 64 + t * 16 + l16][hi * 8];
      v8bf b_lo = *(const v8bf*)pb;          // K = 8*hi + 0..7
      v8bf b_hi = *(const v8bf*)(pb + 16);   // K = 16 + 8*hi + 0..7
#pragma unroll
      for (int j = 0; j < 8; ++j) { bfr[t][j] = b_lo[j]; bfr[t][8 + j] = b_hi[j]; }
    }
#pragma unroll
    for (int tm = 0; tm < 4; ++tm)
#pragma unroll
      for (int tn = 0; tn < 4; ++tn)
        acc[tm][tn] = __builtin_amdgcn_wmma_f32_16x16x32_bf16(
            false, af[tm], false, bfr[tn], (short)0, acc[tm][tn], false, false);
    __syncthreads();
  }

  // epilogue: C layout (i,lane): M = i + 8*hi, N = l16
  for (int tm = 0; tm < 4; ++tm)
    for (int tn = 0; tn < 4; ++tn) {
      int n = n0 + wn * 64 + tn * 16 + l16;
      if (n >= N) continue;
      float bv = bias[n];
#pragma unroll
      for (int i = 0; i < 8; ++i) {
        int m = m0 + wm * 64 + tm * 16 + i + 8 * hi;
        C[(size_t)m * N + n] = acc[tm][tn][i] + bv;
      }
    }
}

// ---------------- b-factor projection: ctx(64) -> bq(128), bk(32), bv(32) ----------
__global__ void bproj_kernel(const float* __restrict__ y1,
                             const float* __restrict__ qbW, const float* __restrict__ qbb,
                             const float* __restrict__ kbW, const float* __restrict__ kbb,
                             const float* __restrict__ vbW, const float* __restrict__ vbb,
                             float* __restrict__ bq, float* __restrict__ bk, float* __restrict__ bv) {
  int idx = blockIdx.x * 256 + threadIdx.x;
  if (idx >= M_ * 192) return;
  int row = idx / 192, j = idx % 192;
  const float* ctx = y1 + (size_t)row * N1_;   // ctx = cols 0..63 (ctxb already fused)
  const float* W; float bias; float* dst; int ld, jj;
  if (j < 128)      { jj = j;       W = qbW; bias = qbb[jj]; dst = bq + (size_t)row * 128 + jj; ld = 128; }
  else if (j < 160) { jj = j - 128; W = kbW; bias = kbb[jj]; dst = bk + (size_t)row * 32 + jj;  ld = 32;  }
  else              { jj = j - 160; W = vbW; bias = vbb[jj]; dst = bv + (size_t)row * 32 + jj;  ld = 32;  }
  float acc = bias;
  for (int f = 0; f < 64; ++f) acc += ctx[f] * W[f * ld + jj];
  *dst = acc;
}

// ---------------- TPA contraction + quantum features ----------------
// grid: B*nh*S blocks, 64 threads. out: [B, nh, S, 16]
__global__ void feat_kernel(const float* __restrict__ y1, int aOff,
                            const float* __restrict__ bfac, int bStride, int nh,
                            const float* __restrict__ qeW, const float* __restrict__ qeb,
                            const float* __restrict__ vqc, const float* __restrict__ ent,
                            float* __restrict__ outF) {
  int gid = blockIdx.x;
  int s = gid % S_;
  int h = (gid / S_) % nh;
  int b = gid / (S_ * nh);
  int row = b * S_ + s;
  __shared__ float q[64];
  __shared__ float f8[8], g8[8];
  int d = threadIdx.x;
  const float* a  = y1 + (size_t)row * N1_ + aOff;
  const float* bf = bfac + (size_t)row * bStride;
  float acc = 0.f;
#pragma unroll
  for (int r = 0; r < R_; ++r) acc += a[r * HD_ + d] * bf[r * nh + h];
  q[d] = acc;
  __syncthreads();
  if (d < NQ_) {
    float ang = qeb[d];
    for (int e = 0; e < HD_; ++e) ang += q[e] * qeW[e * NQ_ + d];
    f8[d] = tanhf(ang) * PI_F;
  }
  __syncthreads();
  float entv = ent[0];
  for (int l = 0; l < NL_; ++l) {
    if (d < NQ_) {
      const float* p = vqc + l * (NQ_ * 3) + d * 3;
      g8[d] = cosf(f8[d] + p[0]) * sinf(f8[d] + p[1]) + cosf(f8[d] + p[2]);
    }
    __syncthreads();
    if (d < NQ_) f8[d] = g8[d] * (1.f + entv * g8[(d + 7) & 7]);   // roll shift=1
    __syncthreads();
  }
  if (d < 16) {
    float nrm = 0.f;
    for (int j = 0; j < NQ_; ++j) {
      float c = cosf(f8[j]), sn = sinf(f8[j]);
      nrm += c * c + sn * sn;
    }
    nrm = sqrtf(nrm) + 1e-6f;
    float val = (d < NQ_) ? cosf(f8[d]) : sinf(f8[d - NQ_]);
    outF[((size_t)(b * nh + h) * S_ + s) * 16 + d] = val / nrm;
  }
}

// ---------------- v contraction: v[b,kv,s,d] = sum_r av*bv ----------------
__global__ void v_kernel(const float* __restrict__ y1, const float* __restrict__ bv,
                         float* __restrict__ vout) {
  int gid = blockIdx.x;
  int s = gid % S_;
  int kv = (gid / S_) % KV_;
  int b = gid / (S_ * KV_);
  int row = b * S_ + s;
  int d = threadIdx.x;
  const float* a  = y1 + (size_t)row * N1_ + 1088;
  const float* bf = bv + (size_t)row * 32;
  float acc = 0.f;
#pragma unroll
  for (int r = 0; r < R_; ++r) acc += a[r * HD_ + d] * bf[r * KV_ + kv];
  vout[((size_t)(b * KV_ + kv) * S_ + s) * HD_ + d] = acc;
}

// ---------------- chunked causal linear attention (WMMA f32 16x16x4) ----------------
// grid: B*H WGs, 128 thr (4 waves, wave w owns head-dim cols [16w,16w+16))
__global__ __launch_bounds__(128) void attn_scan_kernel(
    const float* __restrict__ qf, const float* __restrict__ kf,
    const float* __restrict__ vv, ushort_t* __restrict__ attn_bf) {
  int bid = blockIdx.x;
  int b = bid / H_, h = bid % H_;
  int kvh = h >> 2;    // GQA repeat(4)
  const float* qfp = qf + (size_t)(b * H_ + h) * S_ * 16;
  const float* kfp = kf + (size_t)(b * KV_ + kvh) * S_ * 16;
  const float* vp  = vv + (size_t)(b * KV_ + kvh) * S_ * HD_;
  ushort_t* op = attn_bf + (size_t)b * S_ * E_ + h * HD_;

  __shared__ float sQF[16][16];
  __shared__ float sKF[16][16];
  __shared__ float sV[16][64];
  __shared__ float sKV[16][64];   // state mirror [f][d]
  __shared__ float sKsum[16];
  __shared__ float sMS[16][17];   // masked scores [s][s'], padded
  __shared__ float sDen[16];

  int tid = threadIdx.x;
  int wave = tid >> 5, lane = tid & 31;
  int l16 = lane & 15, hi = lane >> 4;
  int dbase = wave * 16;

  for (int i = tid; i < 16 * 64; i += 128) ((float*)sKV)[i] = 0.f;
  if (tid < 16) sKsum[tid] = 0.f;
  v8f kvc = {};   // wave's KV-state slice in accumulator layout
  __syncthreads();

  for (int c = 0; c < S_ / 16; ++c) {
    const float* q0 = qfp + (size_t)c * 16 * 16;
    const float* k0 = kfp + (size_t)c * 16 * 16;
    const float* v0 = vp  + (size_t)c * 16 * 64;
    for (int i = tid; i < 256; i += 128) { ((float*)sQF)[i] = q0[i]; ((float*)sKF)[i] = k0[i]; }
    for (int i = tid; i < 1024; i += 128) ((float*)sV)[i] = v0[i];
    __syncthreads();

    if (wave == 0) {
      // scores S = qf @ kf^T  (A[m=s][k=f], B[k=f][n=s'] = kf[s'][f])
      v8f sc = {};
#pragma unroll
      for (int kk = 0; kk < 4; ++kk) {
        int k = kk * 4 + hi * 2;
        v2f a;  a.x = sQF[l16][k]; a.y = sQF[l16][k + 1];
        v2f bb; bb.x = sKF[l16][k]; bb.y = sKF[l16][k + 1];
        sc = __builtin_amdgcn_wmma_f32_16x16x4_f32(false, a, false, bb, (short)0, sc, false, false);
      }
      float rs[8];
#pragma unroll
      for (int i = 0; i < 8; ++i) {
        int m = i + 8 * hi;
        float vm = (l16 <= m) ? sc[i] : 0.f;   // causal mask (inclusive)
        sMS[m][l16] = vm;
        float r = vm;
        r += __shfl_xor(r, 1, 16);
        r += __shfl_xor(r, 2, 16);
        r += __shfl_xor(r, 4, 16);
        r += __shfl_xor(r, 8, 16);
        rs[i] = r;
      }
      if (l16 == 0) {
#pragma unroll
        for (int i = 0; i < 8; ++i) {
          int m = i + 8 * hi;
          float dot = 0.f;
          for (int f = 0; f < 16; ++f) dot += sQF[m][f] * sKsum[f];
          sDen[m] = rs[i] + dot;
        }
      }
    }
    __syncthreads();

    // num = qf @ KV_prev + maskedS @ v   (our 16 head-dim cols)
    v8f num = {};
#pragma unroll
    for (int kk = 0; kk < 4; ++kk) {
      int k = kk * 4 + hi * 2;
      v2f a;  a.x = sQF[l16][k]; a.y = sQF[l16][k + 1];
      v2f bb; bb.x = sKV[k][dbase + l16]; bb.y = sKV[k + 1][dbase + l16];
      num = __builtin_amdgcn_wmma_f32_16x16x4_f32(false, a, false, bb, (short)0, num, false, false);
    }
#pragma unroll
    for (int kk = 0; kk < 4; ++kk) {
      int k = kk * 4 + hi * 2;
      v2f a;  a.x = sMS[l16][k]; a.y = sMS[l16][k + 1];
      v2f bb; bb.x = sV[k][dbase + l16]; bb.y = sV[k + 1][dbase + l16];
      num = __builtin_amdgcn_wmma_f32_16x16x4_f32(false, a, false, bb, (short)0, num, false, false);
    }
#pragma unroll
    for (int i = 0; i < 8; ++i) {
      int m = i + 8 * hi;
      int sg = c * 16 + m;
      float val = num[i] / (sDen[m] + 1e-6f);
      op[(size_t)sg * E_ + dbase + l16] = f2bf(val);   // attn output straight to bf16
    }
    // state update: KV += kf^T @ v  (A[m=f][k=s'] = kf[s'][f])
#pragma unroll
    for (int kk = 0; kk < 4; ++kk) {
      int k = kk * 4 + hi * 2;
      v2f a;  a.x = sKF[k][l16]; a.y = sKF[k + 1][l16];
      v2f bb; bb.x = sV[k][dbase + l16]; bb.y = sV[k + 1][dbase + l16];
      kvc = __builtin_amdgcn_wmma_f32_16x16x4_f32(false, a, false, bb, (short)0, kvc, false, false);
    }
#pragma unroll
    for (int i = 0; i < 8; ++i) sKV[i + 8 * hi][dbase + l16] = kvc[i];
    if (wave == 0 && lane < 16) {
      float s = 0.f;
      for (int sp = 0; sp < 16; ++sp) s += sKF[sp][lane];
      sKsum[lane] += s;
    }
    __syncthreads();
  }
}

// ---------------- residual + layernorm ----------------
__global__ __launch_bounds__(256) void ln_kernel(
    const float* __restrict__ x, const float* __restrict__ opre,
    const float* __restrict__ gamma, const float* __restrict__ beta,
    float* __restrict__ out) {
  int m = blockIdx.x;
  int t = threadIdx.x;
  __shared__ float red[256];
  const float* xr = x + (size_t)m * E_;
  const float* orr = opre + (size_t)m * E_;
  float y[4]; float s = 0.f;
#pragma unroll
  for (int i = 0; i < 4; ++i) { int e = t + i * 256; y[i] = xr[e] + orr[e]; s += y[i]; }
  red[t] = s; __syncthreads();
  for (int st = 128; st > 0; st >>= 1) { if (t < st) red[t] += red[t + st]; __syncthreads(); }
  float mu = red[0] * (1.f / E_);
  __syncthreads();
  float vs = 0.f;
#pragma unroll
  for (int i = 0; i < 4; ++i) { float dd = y[i] - mu; vs += dd * dd; }
  red[t] = vs; __syncthreads();
  for (int st = 128; st > 0; st >>= 1) { if (t < st) red[t] += red[t + st]; __syncthreads(); }
  float rstd = rsqrtf(red[0] * (1.f / E_) + 1e-5f);
#pragma unroll
  for (int i = 0; i < 4; ++i) {
    int e = t + i * 256;
    out[(size_t)m * E_ + e] = gamma[e] * (y[i] - mu) * rstd + beta[e];
  }
}

// ---------------- launcher ----------------
extern "C" void kernel_launch(void* const* d_in, const int* in_sizes, int n_in,
                              void* d_out, int out_size, void* d_ws, size_t ws_size,
                              hipStream_t stream) {
  (void)in_sizes; (void)n_in; (void)out_size; (void)ws_size;
  const float* x    = (const float*)d_in[0];
  const float* ctxW = (const float*)d_in[1];
  const float* ctxb = (const float*)d_in[2];
  const float* qaW  = (const float*)d_in[3];
  const float* qab  = (const float*)d_in[4];
  const float* qbW  = (const float*)d_in[5];
  const float* qbb  = (const float*)d_in[6];
  const float* kaW  = (const float*)d_in[7];
  const float* kab  = (const float*)d_in[8];
  const float* kbW  = (const float*)d_in[9];
  const float* kbb  = (const float*)d_in[10];
  const float* vaW  = (const float*)d_in[11];
  const float* vab  = (const float*)d_in[12];
  const float* vbW  = (const float*)d_in[13];
  const float* vbb  = (const float*)d_in[14];
  const float* qeW  = (const float*)d_in[15];
  const float* qeb  = (const float*)d_in[16];
  const float* vqc  = (const float*)d_in[17];
  const float* ent  = (const float*)d_in[18];
  const float* outW = (const float*)d_in[19];
  const float* outb = (const float*)d_in[20];
  const float* gamma= (const float*)d_in[21];
  const float* beta = (const float*)d_in[22];
  float* out = (float*)d_out;

  char* ws = (char*)d_ws;
  size_t off = 0;
  auto take = [&](size_t bytes) -> char* {
    char* p = ws + off;
    off += (bytes + 255) & ~(size_t)255;
    return p;
  };
  ushort_t* xb      = (ushort_t*)take((size_t)M_ * E_ * 2);
  ushort_t* w1b     = (ushort_t*)take((size_t)E_ * N1_ * 2);
  ushort_t* outWb   = (ushort_t*)take((size_t)E_ * E_ * 2);
  float*    bias1   = (float*)   take((size_t)N1_ * 4);
  float*    y1      = (float*)   take((size_t)M_ * N1_ * 4);
  float*    bq      = (float*)   take((size_t)M_ * 128 * 4);
  float*    bk      = (float*)   take((size_t)M_ * 32 * 4);
  float*    bv      = (float*)   take((size_t)M_ * 32 * 4);
  float*    qfbuf   = (float*)   take((size_t)B_ * H_ * S_ * 16 * 4);
  float*    kfbuf   = (float*)   take((size_t)B_ * KV_ * S_ * 16 * 4);
  float*    vbuf    = (float*)   take((size_t)B_ * KV_ * S_ * HD_ * 4);
  ushort_t* attn_bf = (ushort_t*)take((size_t)M_ * E_ * 2);
  float*    out_pre = (float*)   take((size_t)M_ * E_ * 4);

  // 1) pack activations & weights to bf16
  cvt_bf16_kernel<<<(M_ * E_ + 255) / 256, 256, 0, stream>>>(x, xb, M_ * E_);
  cvt_cols_kernel<<<(E_ * CD_ + 255) / 256, 256, 0, stream>>>(ctxW, w1b, E_, CD_, N1_, 0);
  cvt_cols_kernel<<<(E_ * 512 + 255) / 256, 256, 0, stream>>>(qaW, w1b, E_, 512, N1_, 64);
  cvt_cols_kernel<<<(E_ * 512 + 255) / 256, 256, 0, stream>>>(kaW, w1b, E_, 512, N1_, 576);
  cvt_cols_kernel<<<(E_ * 512 + 255) / 256, 256, 0, stream>>>(vaW, w1b, E_, 512, N1_, 1088);
  cvt_cols_kernel<<<(E_ * E_ + 255) / 256, 256, 0, stream>>>(outW, outWb, E_, E_, E_, 0);
  pack_bias1_kernel<<<(N1_ + 255) / 256, 256, 0, stream>>>(ctxb, qab, kab, vab, bias1);

  // 2) fused projection GEMM: y1 = x @ [ctxW|qaW|kaW|vaW] + bias  (bf16 WMMA, TDM-staged A)
  gemm_bf16_kernel<<<dim3(M_ / 128, (N1_ + 255) / 256), 256, 0, stream>>>(
      xb, w1b, bias1, y1, M_, N1_, E_);

  // 3) b-factor projections from ctx
  bproj_kernel<<<(M_ * 192 + 255) / 256, 256, 0, stream>>>(
      y1, qbW, qbb, kbW, kbb, vbW, vbb, bq, bk, bv);

  // 4) TPA contraction + quantum features (q heads and kv heads), v contraction
  feat_kernel<<<B_ * H_ * S_, 64, 0, stream>>>(y1, 64, bq, 128, H_, qeW, qeb, vqc, ent, qfbuf);
  feat_kernel<<<B_ * KV_ * S_, 64, 0, stream>>>(y1, 576, bk, 32, KV_, qeW, qeb, vqc, ent, kfbuf);
  v_kernel<<<B_ * KV_ * S_, 64, 0, stream>>>(y1, bv, vbuf);

  // 5) chunked causal linear attention (f32 WMMA scan), writes attn as bf16
  attn_scan_kernel<<<B_ * H_, 128, 0, stream>>>(qfbuf, kfbuf, vbuf, attn_bf);

  // 6) output projection GEMM (+outb) then residual + layernorm
  gemm_bf16_kernel<<<dim3(M_ / 128, E_ / 256), 256, 0, stream>>>(
      attn_bf, outWb, outb, out_pre, M_, E_, E_);
  ln_kernel<<<M_, 256, 0, stream>>>(x, out_pre, gamma, beta, out);
}
